// LiveSAL_81338090652048
// MI455X (gfx1250) — compile-verified, matrix operand order checked by reading, not run on hardware
//
#include <hip/hip_runtime.h>
#include <hip/hip_bf16.h>

typedef __attribute__((ext_vector_type(16))) _Float16 v16h;
typedef __attribute__((ext_vector_type(2)))  _Float16 h2;
typedef __attribute__((ext_vector_type(8)))  float    v8f;

namespace {

constexpr int  HWp  = 121;                      // 11*11 pixels
constexpr int  LDP  = 34;                       // gemm LDS row stride (halves)
constexpr int  LDPC = 66;                       // conv LDS row stride (halves, K=64)
constexpr long long SLOT = 128LL * 256 * HWp;   // one slot (B,C,H,W) floats
constexpr long long FULL = 5 * SLOT;            // full state (S,B,C,H,W)

// ---------------------------------------------------------------------------
// WMMA fragment load from an LDS tile stored row-major [16][stride] (f16).
// Works for A (rows=M) and B^T (rows=N) per the CDNA5 16-bit 16x32 layout.
// ---------------------------------------------------------------------------
__device__ inline v16h frag_ld(const _Float16* t, int stride) {
  int l  = threadIdx.x & 31;
  int r0 = (l < 16) ? l : (l - 16);
  int kb = (l < 16) ? 0 : 8;
  v16h f;
#pragma unroll
  for (int r = 0; r < 8; ++r) {
    int kk = (r < 4) ? (kb + 2 * r) : (16 + kb + 2 * (r - 4));
    f[2 * r]     = t[r0 * stride + kk];
    f[2 * r + 1] = t[r0 * stride + kk + 1];
  }
  return f;
}

__device__ inline h2 pack2(float a, float b) {
  h2 p;
  p[0] = (_Float16)a;
  p[1] = (_Float16)b;
  return p;
}

// ---------------------------------------------------------------------------
// Implicit-GEMM 3x3 conv, pad=1, stride=1, NCHW, 11x11 spatial.
// grid = (Cout/16, Nsamples), block = 256 (8 waves).
// Block: 16 output channels x 128 pixels; wave w owns pixel tile w.
// K loop: 9 taps x ceil(Cin/64); 2 WMMAs per barrier pair.
// AL=true: Cin % 64 == 0 -> no channel clamps/masks in the hot path.
// ---------------------------------------------------------------------------
template <bool AL>
__global__ __launch_bounds__(256) void k_conv3x3(
    const float* __restrict__ in, const float* __restrict__ w,
    const float* __restrict__ bias, float* __restrict__ out,
    int Cin, int Cout)
{
  const int co0  = blockIdx.x * 16;
  const int smp  = blockIdx.y;
  const int tid  = threadIdx.x;
  const int wid  = tid >> 5;
  const int lane = tid & 31;

  // B staging: half-wave sub covers kk = sub*32 + c; 16 lanes = 16 pixels.
  const int bn   = lane & 15;
  const int bsub = lane >> 4;
  const int bp   = wid * 16 + bn;          // 0..127, this lane's pixel
  const int bpy  = bp / 11, bpx = bp % 11;

  // A staging: thread owns 4 consecutive kk of one row.
  const int am_  = (tid * 4) >> 6;         // row 0..15
  const int akk0 = (tid * 4) & 63;         // first kk (multiple of 4)

  const long long inBase = (long long)smp * Cin * HWp;

  __shared__ _Float16 lA[16 * LDPC];
  __shared__ _Float16 lB[8][16 * LDPC];

  v8f acc = {};

  for (int tap = 0; tap < 9; ++tap) {
    const int ky = tap / 3 - 1;
    const int kx = tap % 3 - 1;
    const int d  = ky * 11 + kx;
    const bool rowok = (bp < HWp) && ((unsigned)(bpy + ky) < 11u) &&
                       ((unsigned)(bpx + kx) < 11u);
    const float bmask  = rowok ? 1.f : 0.f;
    const float* inTap = in + inBase + (rowok ? (bp + d) : 0);
    const float* wRow  = w + (long long)(co0 + am_) * Cin * 9 + tap;

    for (int c0 = 0; c0 < Cin; c0 += 64) {
      // ---- gather phase (no LDS): unconditional loads, mask by multiply ----
      float av[4];
#pragma unroll
      for (int t = 0; t < 4; ++t) {
        int ci = c0 + akk0 + t;
        if (AL) {
          av[t] = wRow[(long long)ci * 9];
        } else {
          int cic = (ci < Cin) ? ci : 0;
          float cm = (ci < Cin) ? 1.f : 0.f;
          av[t] = wRow[(long long)cic * 9] * cm;
        }
      }
      float bv[32];
#pragma unroll
      for (int c = 0; c < 32; ++c) {
        int ci = c0 + bsub * 32 + c;
        if (AL) {
          bv[c] = inTap[(long long)ci * HWp] * bmask;
        } else {
          int cic = (ci < Cin) ? ci : 0;
          float cm = (ci < Cin) ? bmask : 0.f;
          bv[c] = inTap[(long long)cic * HWp] * cm;
        }
      }
      __syncthreads();          // previous chunk's fragment reads complete
      // ---- store phase: packed f16 pairs (b32 LDS stores) ----
#pragma unroll
      for (int t = 0; t < 4; t += 2)
        *(h2*)&lA[am_ * LDPC + akk0 + t] = pack2(av[t], av[t + 1]);
      _Float16* bdst = &lB[wid][bn * LDPC + bsub * 32];
#pragma unroll
      for (int c = 0; c < 32; c += 2)
        *(h2*)&bdst[c] = pack2(bv[c], bv[c + 1]);
      __syncthreads();
      // ---- 2 x WMMA over the 64-wide K slab ----
      v16h a0 = frag_ld(lA, LDPC);
      v16h b0 = frag_ld(lB[wid], LDPC);
      acc = __builtin_amdgcn_wmma_f32_16x16x32_f16(false, a0, false, b0,
                                                   (short)0, acc, false, false);
      v16h a1 = frag_ld(lA + 32, LDPC);
      v16h b1 = frag_ld(lB[wid] + 32, LDPC);
      acc = __builtin_amdgcn_wmma_f32_16x16x32_f16(false, a1, false, b1,
                                                   (short)0, acc, false, false);
    }
  }

  const int nloc = lane & 15;
  const int mb   = (lane < 16) ? 0 : 8;
  const int p    = wid * 16 + nloc;
  if (p < HWp) {
    float* op = out + ((long long)smp * Cout + co0 + mb) * HWp + p;
    const float* bp_ = bias + co0 + mb;
#pragma unroll
    for (int r = 0; r < 8; ++r)
      op[r * HWp] = acc[r] + bp_[r];
  }
}

// ---------------------------------------------------------------------------
// Generic batched strided GEMM, element strides as template constants:
//   D[m][n] = alpha * sum_k A[m][k]*B[k][n]
//   A[m][k] = A[bt*batA + m*ASM + k*ASK]   (f32 -> f16 staged, branchless)
//   B[k][n] = B[bt*batB + k*BSK + n*BSN]
//   D[m][n] = D[bt*batD + m*OSM + n*OSN]
// grid = (ceil(N/128), ceil(M/16), batch), block = 256 (8 waves).
// ---------------------------------------------------------------------------
template <int ASM, int ASK, int BSK, int BSN, int OSM, int OSN>
__global__ __launch_bounds__(256) void k_gemm(
    const float* __restrict__ A, const float* __restrict__ B,
    float* __restrict__ D,
    long long batA, long long batB, long long batD,
    int M, int N, int K, float alpha)
{
  const int bt   = blockIdx.z;
  const int m0   = blockIdx.y * 16;
  const int n0b  = blockIdx.x * 128;
  const int tid  = threadIdx.x;
  const int wid  = tid >> 5;
  const int lane = tid & 31;

  const float* Ab = A + (long long)bt * batA;
  const float* Bb = B + (long long)bt * batB;
  float*       Db = D + (long long)bt * batD;

  const int bn   = lane & 15;
  const int bsub = lane >> 4;

  // Hoisted per-thread row/column pointers + masks.
  const int am0 = (tid * 2) >> 5;                 // A row this thread stages
  const int ak0 = (tid * 2) & 31;                 // two consecutive kk (even)
  const int mgS = m0 + am0;
  const int mgC = (mgS < M) ? mgS : (M - 1);
  const float* aRow  = Ab + (long long)mgC * ASM;
  const float  amask = (mgS < M) ? 1.f : 0.f;

  const int ngl = n0b + wid * 16 + bn;            // this lane's N column
  const int ngc = (ngl < N) ? ngl : (N - 1);
  const float* bRow  = Bb + (long long)ngc * BSN;
  const float  nmask = (ngl < N) ? 1.f : 0.f;

  __shared__ _Float16 lA[16 * LDP];
  __shared__ _Float16 lB[8][16 * LDP];

  v8f acc = {};

  for (int k0 = 0; k0 < K; k0 += 32) {
    float av[2];
#pragma unroll
    for (int t = 0; t < 2; ++t) {
      int kg  = k0 + ak0 + t;
      int kgc = (kg < K) ? kg : (K - 1);
      float cm = (kg < K) ? amask : 0.f;
      av[t] = aRow[(long long)kgc * ASK] * cm;
    }
    float bv[16];
#pragma unroll
    for (int c = 0; c < 16; ++c) {
      int kg  = k0 + bsub * 16 + c;
      int kgc = (kg < K) ? kg : (K - 1);
      float cm = (kg < K) ? nmask : 0.f;
      bv[c] = bRow[(long long)kgc * BSK] * cm;
    }
    __syncthreads();
    *(h2*)&lA[am0 * LDP + ak0] = pack2(av[0], av[1]);
    _Float16* bd = &lB[wid][bn * LDP + bsub * 16];
#pragma unroll
    for (int c = 0; c < 16; c += 2)
      *(h2*)&bd[c] = pack2(bv[c], bv[c + 1]);
    __syncthreads();
    v16h a = frag_ld(lA, LDP);
    v16h b = frag_ld(lB[wid], LDP);
    acc = __builtin_amdgcn_wmma_f32_16x16x32_f16(false, a, false, b,
                                                 (short)0, acc, false, false);
  }

  const int nloc = lane & 15;
  const int mb   = (lane < 16) ? 0 : 8;
  const int ng   = n0b + wid * 16 + nloc;
  if (ng < N) {
    float* dp = Db + (long long)(m0 + mb) * OSM + (long long)ng * OSN;
    if (m0 + 16 <= M) {          // block-uniform full-tile fast path
#pragma unroll
      for (int r = 0; r < 8; ++r)
        dp[r * OSM] = alpha * acc[r];
    } else {
#pragma unroll
      for (int r = 0; r < 8; ++r)
        if (m0 + mb + r < M) dp[r * OSM] = alpha * acc[r];
    }
  }
}

// ---------------------------------------------------------------------------
// LayerNorm over (C,H,W) per sample, gamma/beta shaped (C,H,W), optional ReLU.
// ---------------------------------------------------------------------------
__global__ __launch_bounds__(256) void k_ln(
    const float* __restrict__ in, const float* __restrict__ g,
    const float* __restrict__ b, float* __restrict__ out, int relu)
{
  const int n = 256 * HWp;
  const long long base = (long long)blockIdx.x * n;
  __shared__ float s1[256];
  __shared__ float s2[256];
  float a1 = 0.f, a2 = 0.f;
  for (int i = threadIdx.x; i < n; i += 256) {
    float v = in[base + i];
    a1 += v; a2 += v * v;
  }
  s1[threadIdx.x] = a1; s2[threadIdx.x] = a2;
  __syncthreads();
  for (int st = 128; st > 0; st >>= 1) {
    if (threadIdx.x < st) {
      s1[threadIdx.x] += s1[threadIdx.x + st];
      s2[threadIdx.x] += s2[threadIdx.x + st];
    }
    __syncthreads();
  }
  float mean = s1[0] / n;
  float var  = s2[0] / n - mean * mean;
  float inv  = rsqrtf(var + 1e-5f);
  for (int i = threadIdx.x; i < n; i += 256) {
    float v = (in[base + i] - mean) * inv * g[i] + b[i];
    if (relu) v = fmaxf(v, 0.f);
    out[base + i] = v;
  }
}

// Row softmax over 121 valid cols of a 128-strided score row; pads -> 0.
__global__ __launch_bounds__(128) void k_softmax(float* __restrict__ sc)
{
  const long long base = ((long long)blockIdx.y * HWp + blockIdx.x) * 128;
  const int t = threadIdx.x;
  __shared__ float red[128];
  float v = (t < HWp) ? sc[base + t] : -1e30f;
  red[t] = v; __syncthreads();
  for (int st = 64; st > 0; st >>= 1) {
    if (t < st) red[t] = fmaxf(red[t], red[t + st]);
    __syncthreads();
  }
  float mx = red[0];
  __syncthreads();
  float e = (t < HWp) ? __expf(v - mx) : 0.f;
  red[t] = e; __syncthreads();
  for (int st = 64; st > 0; st >>= 1) {
    if (t < st) red[t] += red[t + st];
    __syncthreads();
  }
  sc[base + t] = e / red[0];
}

__global__ __launch_bounds__(256) void k_axpby(
    float* __restrict__ out, const float* __restrict__ x,
    const float* __restrict__ y, const float* __restrict__ aPtr,
    int mode, long long n)
{
  long long i = (long long)blockIdx.x * 256 + threadIdx.x;
  if (i >= n) return;
  float a = aPtr[0];
  out[i] = (mode == 0) ? (a * x[i] + y[i]) : (a * x[i] + (1.f - a) * y[i]);
}

__global__ __launch_bounds__(256) void k_zero(float* __restrict__ p, long long n)
{
  long long i = (long long)blockIdx.x * 256 + threadIdx.x;
  if (i < n) p[i] = 0.f;
}

__global__ __launch_bounds__(256) void k_sigmoid(float* __restrict__ p, long long n)
{
  long long i = (long long)blockIdx.x * 256 + threadIdx.x;
  if (i < n) p[i] = 1.f / (1.f + __expf(-p[i]));
}

__global__ __launch_bounds__(256) void k_tanh(float* __restrict__ p, long long n)
{
  long long i = (long long)blockIdx.x * 256 + threadIdx.x;
  if (i < n) p[i] = tanhf(p[i]);
}

// dst(B,259,HW) = concat(h_j, tenc0, tenc1, rpe[idx]) along channels.
__global__ __launch_bounds__(256) void k_build_m(
    float* __restrict__ dst, const float* __restrict__ hj,
    const float* __restrict__ rpeRow, float c0, float c1)
{
  const long long n = 128LL * 259 * HWp;
  long long i = (long long)blockIdx.x * 256 + threadIdx.x;
  if (i >= n) return;
  int p  = (int)(i % HWp);
  int ch = (int)((i / HWp) % 259);
  long long b = i / (259LL * HWp);
  float v;
  if (ch < 256)       v = hj[(b * 256 + ch) * HWp + p];
  else if (ch == 256) v = c0;
  else if (ch == 257) v = c1;
  else                v = rpeRow[p];
  dst[i] = v;
}

// dst(B,512,HW) = concat(a, hprev * r) ; r = zr[:, ch] for ch in [256,512), or 1.
__global__ __launch_bounds__(256) void k_concat_gru(
    float* __restrict__ dst, const float* __restrict__ a,
    const float* __restrict__ hprev, const float* __restrict__ zr)
{
  const long long n = 128LL * 512 * HWp;
  long long i = (long long)blockIdx.x * 256 + threadIdx.x;
  if (i >= n) return;
  int p  = (int)(i % HWp);
  int ch = (int)((i / HWp) % 512);
  long long b = i / (512LL * HWp);
  float v;
  if (ch < 256) {
    v = a[(b * 256 + ch) * HWp + p];
  } else {
    float hv = hprev[(b * 256 + (ch - 256)) * HWp + p];
    float r  = zr ? zr[(b * 512 + ch) * HWp + p] : 1.f;
    v = hv * r;
  }
  dst[i] = v;
}

// nh = (1-z)*hprev + z*hhat + hprev ; z = zr[:, :256].
__global__ __launch_bounds__(256) void k_gru_final(
    float* __restrict__ nh, const float* __restrict__ zr,
    const float* __restrict__ hprev, const float* __restrict__ hhat)
{
  const long long n = SLOT;
  long long i = (long long)blockIdx.x * 256 + threadIdx.x;
  if (i >= n) return;
  int p = (int)(i % HWp);
  int c = (int)((i / HWp) % 256);
  long long b = i / (256LL * HWp);
  float z  = zr[(b * 512 + c) * HWp + p];
  float hp = hprev[i];
  nh[i] = (1.f - z) * hp + z * hhat[i] + hp;
}

// means(B,512) = spatial mean of concat(xn_e, msg).
__global__ __launch_bounds__(256) void k_mean2(
    float* __restrict__ means, const float* __restrict__ xne,
    const float* __restrict__ msg)
{
  const long long n = 128LL * 512;
  long long i = (long long)blockIdx.x * 256 + threadIdx.x;
  if (i >= n) return;
  int ci = (int)(i % 512);
  long long b = i / 512;
  const float* src = (ci < 256) ? (xne + (b * 256 + ci) * HWp)
                                : (msg + (b * 256 + (ci - 256)) * HWp);
  float s = 0.f;
  for (int p = 0; p < HWp; ++p) s += src[p];
  means[i] = s * (1.f / HWp);
}

// gates(B,256) = sigmoid(gw @ means + gb)  (1x1 conv + spatial mean commute).
__global__ __launch_bounds__(256) void k_gate(
    float* __restrict__ gates, const float* __restrict__ means,
    const float* __restrict__ gw, const float* __restrict__ gb)
{
  const long long n = 128LL * 256;
  long long i = (long long)blockIdx.x * 256 + threadIdx.x;
  if (i >= n) return;
  int co = (int)(i % 256);
  long long b = i / 256;
  float s = gb[co];
  for (int ci = 0; ci < 512; ++ci)
    s += gw[co * 512 + ci] * means[b * 512 + ci];
  gates[i] = 1.f / (1.f + __expf(-s));
}

// acc += msg * gates[b][c]
__global__ __launch_bounds__(256) void k_gate_acc(
    float* __restrict__ acc, const float* __restrict__ msg,
    const float* __restrict__ gates)
{
  const long long n = SLOT;
  long long i = (long long)blockIdx.x * 256 + threadIdx.x;
  if (i >= n) return;
  int c = (int)((i / HWp) % 256);
  long long b = i / (256LL * HWp);
  acc[i] += msg[i] * gates[b * 256 + c];
}

inline unsigned nb256(long long n) { return (unsigned)((n + 255) / 256); }

} // namespace

// ---------------------------------------------------------------------------
// Host orchestration (full NITER=2 forward).
// ---------------------------------------------------------------------------
extern "C" void kernel_launch(void* const* d_in, const int* in_sizes, int n_in,
                              void* d_out, int out_size, void* d_ws, size_t ws_size,
                              hipStream_t stream)
{
  (void)in_sizes; (void)n_in; (void)ws_size;
  const float* x    = (const float*)d_in[0];
  const float* ing  = (const float*)d_in[1];
  const float* inb  = (const float*)d_in[2];
  const float* iqw  = (const float*)d_in[3];
  const float* iqb  = (const float*)d_in[4];
  const float* ikw  = (const float*)d_in[5];
  const float* ikb  = (const float*)d_in[6];
  const float* ivw  = (const float*)d_in[7];
  const float* ivb  = (const float*)d_in[8];
  const float* iow  = (const float*)d_in[9];
  const float* iob  = (const float*)d_in[10];
  const float* iong = (const float*)d_in[11];
  const float* ionb = (const float*)d_in[12];
  const float* ia   = (const float*)d_in[13];
  const float* eng  = (const float*)d_in[14];
  const float* enb  = (const float*)d_in[15];
  const float* rpe  = (const float*)d_in[16];
  const float* mw   = (const float*)d_in[17];
  const float* mb   = (const float*)d_in[18];
  const float* eqw  = (const float*)d_in[19];
  const float* eqb  = (const float*)d_in[20];
  const float* ekw  = (const float*)d_in[21];
  const float* ekb  = (const float*)d_in[22];
  const float* evw  = (const float*)d_in[23];
  const float* evb  = (const float*)d_in[24];
  const float* gw   = (const float*)d_in[25];
  const float* gb   = (const float*)d_in[26];
  const float* eow  = (const float*)d_in[27];
  const float* eob  = (const float*)d_in[28];
  const float* eong = (const float*)d_in[29];
  const float* eonb = (const float*)d_in[30];
  const float* iia  = (const float*)d_in[31];
  const float* zrw  = (const float*)d_in[32];
  const float* zrb  = (const float*)d_in[33];
  const float* hw_  = (const float*)d_in[34];
  const float* hb_  = (const float*)d_in[35];
  const float* ng   = (const float*)d_in[36];
  const float* nb   = (const float*)d_in[37];

  const float SCALE = 0.0625f;   // 256^-0.5

  // Workspace carve-up (floats). Total ~238M floats (~0.95 GB).
  float* W = (float*)d_ws;
  long long off = 0;
  auto alloc = [&](long long nf) { float* p = W + off; off += nf; return p; };
  float* hA   = alloc(FULL);
  float* hB   = alloc(FULL);
  float* xn   = alloc(FULL);
  float* qb   = alloc(FULL);
  float* kb   = alloc(FULL);
  float* vb   = alloc(FULL);
  float* ob   = alloc(FULL);
  float* hib  = alloc(FULL);
  float* sc   = alloc(1024LL * HWp * 128);
  float* xne  = alloc(SLOT);
  float* qe   = alloc(SLOT);
  float* ybuf = alloc(SLOT);
  float* keb  = alloc(SLOT);
  float* veb  = alloc(SLOT);
  float* msg  = alloc(SLOT);
  float* t1b  = alloc(SLOT);
  float* accb = alloc(SLOT);
  float* comb = alloc(SLOT);
  float* hh   = alloc(SLOT);
  float* nh   = alloc(SLOT);
  float* mbuf = alloc(128LL * 259 * HWp);
  float* cat  = alloc(2 * SLOT);
  float* zr   = alloc(2 * SLOT);
  float* means = alloc(128LL * 512);
  float* gates = alloc(128LL * 256);

  auto conv = [&](const float* in_, const float* w_, const float* b_,
                  float* out_, int Cin, int Cout, int Ns) {
    if ((Cin & 63) == 0)
      k_conv3x3<true><<<dim3(Cout / 16, Ns), 256, 0, stream>>>(
          in_, w_, b_, out_, Cin, Cout);
    else
      k_conv3x3<false><<<dim3(Cout / 16, Ns), 256, 0, stream>>>(
          in_, w_, b_, out_, Cin, Cout);
  };

  hipMemcpyAsync(hA, x, (size_t)FULL * sizeof(float),
                 hipMemcpyDeviceToDevice, stream);

  float* hcur = hA;
  float* hnew = hB;

  for (int iter = 0; iter < 2; ++iter) {
    // ---------------- intra_all over all S*B samples ----------------
    k_ln<<<640, 256, 0, stream>>>(hcur, ing, inb, xn, 0);
    conv(xn, iqw, iqb, qb, 256, 256, 640);
    conv(xn, ikw, ikb, kb, 256, 256, 640);
    conv(xn, ivw, ivb, vb, 256, 256, 640);
    // scores[n][m] = SCALE * sum_c q[c,n]k[c,m]  (batch 640, K=256)
    k_gemm<1, HWp, HWp, 1, 128, 1><<<dim3(1, 8, 640), 256, 0, stream>>>(
        qb, kb, sc, 30976, 30976, 15488, HWp, HWp, 256, SCALE);
    k_softmax<<<dim3(HWp, 640), 128, 0, stream>>>(sc);
    // o[c,n] = sum_m a[n,m] v[c,m]  (M=121 rows n, N=256 cols c, K=121)
    k_gemm<128, 1, 1, HWp, 1, HWp><<<dim3(2, 8, 640), 256, 0, stream>>>(
        sc, vb, ob, 15488, 30976, 30976, HWp, 256, HWp, 1.f);
    conv(ob, iow, iob, qb, 256, 256, 640);          // reuse qb as t1
    k_ln<<<640, 256, 0, stream>>>(qb, iong, ionb, kb, 1);  // relu(ln) -> kb
    k_axpby<<<nb256(FULL), 256, 0, stream>>>(hib, kb, xn, ia, 0, FULL);

    // ---------------- inter + GRU per slot ----------------
    for (int i = 0; i < 5; ++i) {
      const float* h_i  = hcur + (long long)i * SLOT;
      const float* hi_i = hib  + (long long)i * SLOT;

      k_ln<<<128, 256, 0, stream>>>(h_i, eng, enb, xne, 0);
      conv(xne, eqw, eqb, qe, 256, 256, 128);
      k_zero<<<nb256(SLOT), 256, 0, stream>>>(accb, SLOT);

      for (int j = 0; j < 5; ++j) {
        int dd = i - j;
        if (j == i || (dd > 2 || dd < -2)) continue;
        int idx = ((dd + 1) % 4 + 4) % 4;
        float c0 = (i < j) ? 1.f : -1.f;
        float c1 = (float)(dd < 0 ? -dd : dd) / 2.f;

        k_build_m<<<nb256(128LL * 259 * HWp), 256, 0, stream>>>(
            mbuf, hcur + (long long)j * SLOT, rpe + (long long)idx * HWp, c0, c1);
        conv(mbuf, mw, mb, ybuf, 259, 256, 128);
        conv(ybuf, ekw, ekb, keb, 256, 256, 128);
        conv(ybuf, evw, evb, veb, 256, 256, 128);
        // per-head scores: batch B*HEADS=1024, K=HD=32
        k_gemm<1, HWp, HWp, 1, 128, 1><<<dim3(1, 8, 1024), 256, 0, stream>>>(
            qe, keb, sc, 3872, 3872, 15488, HWp, HWp, 32, SCALE);
        k_softmax<<<dim3(HWp, 1024), 128, 0, stream>>>(sc);
        // msg[d][n] = sum_m v[d,m] a[n,m]  (M=32, N=121, K=121)
        k_gemm<HWp, 1, 1, 128, HWp, 1><<<dim3(1, 2, 1024), 256, 0, stream>>>(
            veb, sc, msg, 3872, 15488, 3872, 32, HWp, HWp, 1.f);
        conv(msg, eow, eob, t1b, 256, 256, 128);
        k_ln<<<128, 256, 0, stream>>>(t1b, eong, eonb, msg, 1);  // relu(ln)
        k_mean2<<<nb256(128LL * 512), 256, 0, stream>>>(means, xne, msg);
        k_gate<<<nb256(128LL * 256), 256, 0, stream>>>(gates, means, gw, gb);
        k_gate_acc<<<nb256(SLOT), 256, 0, stream>>>(accb, msg, gates);
      }

      // comb = iia*hi + (1-iia)*inter
      k_axpby<<<nb256(SLOT), 256, 0, stream>>>(comb, hi_i, accb, iia, 1, SLOT);
      // GRU
      k_concat_gru<<<nb256(2 * SLOT), 256, 0, stream>>>(cat, comb, h_i, nullptr);
      conv(cat, zrw, zrb, zr, 512, 512, 128);
      k_sigmoid<<<nb256(2 * SLOT), 256, 0, stream>>>(zr, 2 * SLOT);
      k_concat_gru<<<nb256(2 * SLOT), 256, 0, stream>>>(cat, comb, h_i, zr);
      conv(cat, hw_, hb_, hh, 512, 256, 128);
      k_tanh<<<nb256(SLOT), 256, 0, stream>>>(hh, SLOT);
      k_gru_final<<<nb256(SLOT), 256, 0, stream>>>(nh, zr, h_i, hh);
      k_ln<<<128, 256, 0, stream>>>(nh, ng, nb, hnew + (long long)i * SLOT, 0);
    }

    float* tmp = hcur; hcur = hnew; hnew = tmp;
  }

  hipMemcpyAsync(d_out, hcur, (size_t)out_size * sizeof(float),
                 hipMemcpyDeviceToDevice, stream);
}
